// Performer_85504208928775
// MI455X (gfx1250) — matrix operand order repaired and checked by reading, not compiled, not run on hardware
//
#include <hip/hip_runtime.h>
#include <hip/hip_bf16.h>

// ---------------------------------------------------------------------------
// Performer attention, CDNA5 (gfx1250) wave32 + v_wmma_f32_16x16x32_f16.
// ---------------------------------------------------------------------------

typedef __attribute__((ext_vector_type(16))) _Float16     v16h;
typedef __attribute__((ext_vector_type(8)))  _Float16     h8;
typedef __attribute__((ext_vector_type(8)))  float        v8f;
typedef __attribute__((ext_vector_type(4)))  float        f4;
typedef __attribute__((ext_vector_type(4)))  unsigned int u4;

union FragU { v16h v; u4 u[2]; };

// A-fragment (16x32 f16, MxK): lane = row (lane&15); per-lane halves are
// K = klo..klo+7 and K = 16+klo..16+klo+7 with klo = (lane>>4)*8.
__device__ __forceinline__ v16h frag_a(const _Float16* base, int stride,
                                       int row, int k0, int lane) {
  const _Float16* p = base + row * stride + k0 + ((lane >> 4) << 3);
  FragU f;
  f.u[0] = *(const u4*)p;
  f.u[1] = *(const u4*)(p + 16);
  return f.v;
}

// B-fragment (32x16 f16, KxN): lane = column (lane&15); per-lane halves are
// K = klo..klo+7 and K = 16+klo..16+klo+7 with klo = (lane>>4)*8.
__device__ __forceinline__ v16h frag_b(const _Float16* base, int stride,
                                       int col, int k0, int lane) {
  const _Float16* p = base + col * stride + k0 + ((lane >> 4) << 3);
  FragU f;
  f.u[0] = *(const u4*)p;
  f.u[1] = *(const u4*)(p + 16);
  return f.v;
}

__device__ __forceinline__ v8f wmma_f16(v16h a, v16h b, v8f c) {
  return __builtin_amdgcn_wmma_f32_16x16x32_f16(false, a, false, b,
                                                (short)0, c, false, false);
}

__device__ __forceinline__ v8f zero8() {
  v8f z;
#pragma unroll
  for (int i = 0; i < 8; ++i) z[i] = 0.f;
  return z;
}

// stage 8 elements into LDS as f16 (with conversion for f32 sources)
__device__ __forceinline__ void stage8(const float* s, _Float16* d) {
  f4 a = *(const f4*)s;
  f4 b = *(const f4*)(s + 4);
  h8 h;
#pragma unroll
  for (int i = 0; i < 4; ++i) { h[i] = (_Float16)a[i]; h[i + 4] = (_Float16)b[i]; }
  *(h8*)d = h;
}
__device__ __forceinline__ void stage8(const _Float16* s, _Float16* d) {
  *(h8*)d = *(const h8*)s;
}

// Performer feature-map epilogue value: 0.125*(exp(z)+1e-6).
// __expf -> single v_exp_f32 (vs. libm expf's long VALU sequence).
__device__ __forceinline__ float featmap(float acc, float diag_half) {
  return 0.125f * (__expf(acc - diag_half) + 1e-6f);
}

// ---------------------------------------------------------------------------
// Generic big GEMM: C[M,N] = scale * (A[M,K] @ B[K,N] + bias[N])
//   A: f32 or f16 row-major, B: f32 row-major (converted to f16 while
//   staging to LDS), C: f16 or f32.  M,N multiples of 128; K multiple of 32.
// Block: 256 thr (8 waves); tile 128x128; wave tile 32x64 (2x4 WMMA accs).
// ---------------------------------------------------------------------------
template <typename TA, typename TC>
__global__ __launch_bounds__(256) void gemm_kernel(
    const TA* __restrict__ A, const float* __restrict__ B,
    const float* __restrict__ bias, TC* __restrict__ C,
    int M, int N, int K, int lda, int ldb, int ldc, float scale) {
  const int tid = threadIdx.x, lane = tid & 31, wave = tid >> 5;
  const int wave_m = wave & 3, wave_n = wave >> 2;
  const int row0 = blockIdx.y * 128, col0 = blockIdx.x * 128;

  __shared__ __align__(16) _Float16 As[128 * 40];
  __shared__ __align__(16) _Float16 Bs[128 * 40];

  v8f acc[2][4];
#pragma unroll
  for (int mt = 0; mt < 2; ++mt)
#pragma unroll
    for (int nt = 0; nt < 4; ++nt) acc[mt][nt] = zero8();

  for (int k0 = 0; k0 < K; k0 += 32) {
    // stage A tile: 128 rows x 32 cols -> f16 (coalesced 32B or 16B rows)
#pragma unroll
    for (int i = 0; i < 2; ++i) {
      int slot = tid + i * 256;
      int r = slot >> 2, g = (slot & 3) * 8;
      stage8(A + (size_t)(row0 + r) * lda + k0 + g, &As[r * 40 + g]);
    }
    // stage B tile transposed: read 8 contiguous floats along N
    // (coalesced global_load_b128 x2), scatter f16 into Bs[n][k].
#pragma unroll
    for (int i = 0; i < 2; ++i) {
      int slot = tid + i * 256;
      int k = slot >> 4, ng = (slot & 15) * 8;
      const float* bp = B + (size_t)(k0 + k) * ldb + col0 + ng;
      f4 a = *(const f4*)bp;
      f4 b = *(const f4*)(bp + 4);
#pragma unroll
      for (int ii = 0; ii < 4; ++ii) {
        Bs[(ng + ii) * 40 + k]     = (_Float16)a[ii];
        Bs[(ng + 4 + ii) * 40 + k] = (_Float16)b[ii];
      }
    }
    if (k0 + 32 < K)
      __builtin_prefetch(A + (size_t)(row0 + (tid >> 1)) * lda + k0 + 32, 0, 1);
    __syncthreads();

    v16h af[2], bf[4];
#pragma unroll
    for (int mt = 0; mt < 2; ++mt)
      af[mt] = frag_a(As, 40, wave_m * 32 + mt * 16 + (lane & 15), 0, lane);
#pragma unroll
    for (int nt = 0; nt < 4; ++nt)
      bf[nt] = frag_b(Bs, 40, wave_n * 64 + nt * 16 + (lane & 15), 0, lane);
#pragma unroll
    for (int mt = 0; mt < 2; ++mt)
#pragma unroll
      for (int nt = 0; nt < 4; ++nt)
        acc[mt][nt] = wmma_f16(af[mt], bf[nt], acc[mt][nt]);
    __syncthreads();
  }

#pragma unroll
  for (int nt = 0; nt < 4; ++nt) {
    int n_loc = wave_n * 64 + nt * 16 + (lane & 15);
    float bz = bias[col0 + n_loc];
#pragma unroll
    for (int mt = 0; mt < 2; ++mt) {
#pragma unroll
      for (int j = 0; j < 8; ++j) {
        int m_loc = wave_m * 32 + mt * 16 + ((lane >> 4) << 3) + j;
        float v = scale * (acc[mt][nt][j] + bz);
        C[(size_t)(row0 + m_loc) * ldc + col0 + n_loc] = (TC)v;
      }
    }
  }
}

// ---------------------------------------------------------------------------
// diag[r,h] = sum_d x[r, h*64+d]^2   (x is pre-scaled f16)  -> f32 [16384,16]
// ---------------------------------------------------------------------------
__global__ __launch_bounds__(256) void diag_kernel(const _Float16* __restrict__ x,
                                                   float* __restrict__ dg) {
  int idx = blockIdx.x * 256 + threadIdx.x;  // 16384*16 threads
  int r = idx >> 4, h = idx & 15;
  const _Float16* p = x + (size_t)r * 1024 + h * 64;
  float s = 0.f;
#pragma unroll
  for (int g = 0; g < 8; ++g) {
    h8 hv = *(const h8*)(p + g * 8);
#pragma unroll
    for (int e = 0; e < 8; ++e) { float v = (float)hv[e]; s += v * v; }
  }
  dg[idx] = s;
}

// ---------------------------------------------------------------------------
// kv_kernel: per (b,h) batch, F-tile of 128:
//   k'[n,f] = 0.125*(exp(ks[n,:]@proj[:,f] - diag/2) + 1e-6)   (inner WMMA)
//   kv[f,d] += k'[n,f]*v[n,d]   ksum[f] += k'[n,f]             (outer WMMA)
// grid: (2, 64)  block: 256
// ---------------------------------------------------------------------------
__global__ __launch_bounds__(256) void kv_kernel(
    const _Float16* __restrict__ ks, const _Float16* __restrict__ vh,
    const float* __restrict__ proj, const float* __restrict__ diagk,
    float* __restrict__ kv, float* __restrict__ ksum) {
  const int tid = threadIdx.x, lane = tid & 31, wave = tid >> 5;
  const int bh = blockIdx.y, b = bh >> 4, h = bh & 15;
  const int f0 = blockIdx.x * 128;
  const size_t rowb = (size_t)b * 4096;
  const int col = h * 64;

  __shared__ __align__(16) _Float16 PbT[128 * 72];  // proj^T slice [f][d]
  __shared__ __align__(16) _Float16 Ks[32 * 72];    // ks chunk [n][d]
  __shared__ __align__(16) _Float16 VsT[64 * 40];   // v chunk  [d][n]
  __shared__ __align__(16) _Float16 Ps[128 * 40];   // k' chunk [f][n]
  __shared__ float Ds[32];
  __shared__ float ksl[128];

  {  // stage proj slice once (transposed, f32->f16)
    int f = tid >> 1, d0 = (tid & 1) * 32;
#pragma unroll
    for (int ii = 0; ii < 32; ++ii)
      PbT[f * 72 + d0 + ii] = (_Float16)proj[(size_t)(d0 + ii) * 256 + f0 + f];
  }
  if (tid < 128) ksl[tid] = 0.f;

  float ksacc = 0.f;
  const int wave_f = wave >> 2, wave_d = wave & 3;
  v8f kvacc[4];
#pragma unroll
  for (int mt = 0; mt < 4; ++mt) kvacc[mt] = zero8();

  for (int n0 = 0; n0 < 4096; n0 += 32) {
    {  // stage Ks, VsT, Ds
      int n = tid >> 3, d0 = (tid & 7) * 8;
      const _Float16* kp = ks + (rowb + n0 + n) * 1024 + col + d0;
      *(h8*)&Ks[n * 72 + d0] = *(const h8*)kp;
      h8 hv = *(const h8*)(vh + (rowb + n0 + n) * 1024 + col + d0);
#pragma unroll
      for (int ii = 0; ii < 8; ++ii) VsT[(d0 + ii) * 40 + n] = hv[ii];
      if (tid < 32) Ds[tid] = diagk[(rowb + n0 + tid) * 16 + h];
    }
    __syncthreads();

    // ---- inner: P[32n x 128f] = Ks @ PbT ; wave -> f-tile = wave
    v8f pin[2];
    pin[0] = zero8(); pin[1] = zero8();
#pragma unroll
    for (int ks2 = 0; ks2 < 2; ++ks2) {
      int kk = ks2 * 32;
      v16h bfr = frag_b(PbT, 72, wave * 16 + (lane & 15), kk, lane);
#pragma unroll
      for (int mt = 0; mt < 2; ++mt) {
        v16h afr = frag_a(Ks, 72, mt * 16 + (lane & 15), kk, lane);
        pin[mt] = wmma_f16(afr, bfr, pin[mt]);
      }
    }
    {  // epilogue: exp, store transposed to Ps, accumulate ksum in regs
      int f_loc = wave * 16 + (lane & 15);
#pragma unroll
      for (int mt = 0; mt < 2; ++mt) {
        int nb = mt * 16 + ((lane >> 4) << 3);
#pragma unroll
        for (int j = 0; j < 8; ++j) {
          float val = featmap(pin[mt][j], 0.5f * Ds[nb + j]);
          Ps[f_loc * 40 + nb + j] = (_Float16)val;
          ksacc += val;
        }
      }
    }
    __syncthreads();

    // ---- outer: kv[128f x 64d] += Ps @ VsT^T  (K = 32 n)
    {
      v16h bfr = frag_b(VsT, 40, wave_d * 16 + (lane & 15), 0, lane);
#pragma unroll
      for (int mt = 0; mt < 4; ++mt) {
        v16h afr = frag_a(Ps, 40, wave_f * 64 + mt * 16 + (lane & 15), 0, lane);
        kvacc[mt] = wmma_f16(afr, bfr, kvacc[mt]);
      }
    }
    __syncthreads();
  }

  // write kv (f32)
#pragma unroll
  for (int mt = 0; mt < 4; ++mt) {
    int d_loc = wave_d * 16 + (lane & 15);
#pragma unroll
    for (int j = 0; j < 8; ++j) {
      int f_loc = wave_f * 64 + mt * 16 + ((lane >> 4) << 3) + j;
      kv[((size_t)bh * 256 + f0 + f_loc) * 64 + d_loc] = kvacc[mt][j];
    }
  }
  atomicAdd(&ksl[wave * 16 + (lane & 15)], ksacc);
  __syncthreads();
  if (tid < 128) ksum[(size_t)bh * 256 + f0 + tid] = ksl[tid];
}

// ---------------------------------------------------------------------------
// qkv_kernel: per (b,h), n-tile of 128:
//   q'[n,f] recomputed chunk-wise (inner WMMA + exp),
//   out[n,d] = (q' @ kv) / (q' @ ksum + eps)  (outer WMMA, norm via ds_add)
// grid: (32, 64)  block: 256
// ---------------------------------------------------------------------------
__global__ __launch_bounds__(256) void qkv_kernel(
    const _Float16* __restrict__ qs, const float* __restrict__ kv,
    const float* __restrict__ ksum, const float* __restrict__ proj,
    const float* __restrict__ diagq, _Float16* __restrict__ outp) {
  const int tid = threadIdx.x, lane = tid & 31, wave = tid >> 5;
  const int bh = blockIdx.y, b = bh >> 4, h = bh & 15;
  const int n0 = blockIdx.x * 128;
  const size_t rowb = (size_t)b * 4096;
  const int col = h * 64;

  __shared__ __align__(16) _Float16 Qt[128 * 72];    // qs tile [n][d]
  __shared__ __align__(16) _Float16 kvT[64 * 264];   // kv^T    [d][f]
  __shared__ __align__(16) _Float16 PbT2[32 * 72];   // proj^T chunk [f][d]
  __shared__ __align__(16) _Float16 Qp[128 * 40];    // q' chunk [n][f]
  __shared__ float Ds2[128];
  __shared__ float normS[128];
  __shared__ float ksl2[256];

  // stage Qt (once)
#pragma unroll
  for (int i = 0; i < 4; ++i) {
    int slot = tid + i * 256;
    int n = slot >> 3, d0 = (slot & 7) * 8;
    *(h8*)&Qt[n * 72 + d0] = *(const h8*)(qs + (rowb + n0 + n) * 1024 + col + d0);
  }
  // stage kvT (once, f32 -> f16, transposed)
#pragma unroll
  for (int i = 0; i < 8; ++i) {
    int slot = tid + i * 256;
    int d = slot >> 5, fg = (slot & 31) * 8;
#pragma unroll
    for (int ii = 0; ii < 8; ++ii)
      kvT[d * 264 + fg + ii] =
          (_Float16)kv[((size_t)bh * 256 + fg + ii) * 64 + d];
  }
  if (tid < 128) {
    Ds2[tid] = diagq[(rowb + n0 + tid) * 16 + h];
    normS[tid] = 0.f;
  }
  ksl2[tid] = ksum[(size_t)bh * 256 + tid];

  v8f oacc[4];
#pragma unroll
  for (int nt = 0; nt < 4; ++nt) oacc[nt] = zero8();

  for (int fc0 = 0; fc0 < 256; fc0 += 32) {
    {  // stage proj chunk transposed
      int f = tid >> 3, d0 = (tid & 7) * 8;
#pragma unroll
      for (int ii = 0; ii < 8; ++ii)
        PbT2[f * 72 + d0 + ii] =
            (_Float16)proj[(size_t)(d0 + ii) * 256 + fc0 + f];
    }
    __syncthreads();

    // ---- inner: q'chunk[128n x 32f] = Qt @ PbT2 ; wave -> n-tile = wave
    v8f qin[2];
    qin[0] = zero8(); qin[1] = zero8();
#pragma unroll
    for (int ks2 = 0; ks2 < 2; ++ks2) {
      int kk = ks2 * 32;
      v16h afr = frag_a(Qt, 72, wave * 16 + (lane & 15), kk, lane);
#pragma unroll
      for (int ft = 0; ft < 2; ++ft) {
        v16h bfr = frag_b(PbT2, 72, ft * 16 + (lane & 15), kk, lane);
        qin[ft] = wmma_f16(afr, bfr, qin[ft]);
      }
    }
    {  // epilogue: exp, store Qp row-major, accumulate norm (ds_add_f32)
      int nb = wave * 16 + ((lane >> 4) << 3);
#pragma unroll
      for (int ft = 0; ft < 2; ++ft) {
        int f_loc = ft * 16 + (lane & 15);
        float ksv = ksl2[fc0 + f_loc];
#pragma unroll
        for (int j = 0; j < 8; ++j) {
          float val = featmap(qin[ft][j], 0.5f * Ds2[nb + j]);
          Qp[(nb + j) * 40 + f_loc] = (_Float16)val;
          atomicAdd(&normS[nb + j], val * ksv);
        }
      }
    }
    __syncthreads();

    // ---- outer: out[128n x 64d] += Qp @ kvT^T  (K = 32 f)
    {
      v16h afr = frag_a(Qp, 40, wave * 16 + (lane & 15), 0, lane);
#pragma unroll
      for (int nt = 0; nt < 4; ++nt) {
        v16h bfr = frag_b(kvT, 264, nt * 16 + (lane & 15), fc0, lane);
        oacc[nt] = wmma_f16(afr, bfr, oacc[nt]);
      }
    }
  }
  __syncthreads();

  // divide by norm, write f16 pre-output in [B*N, 1024] layout
#pragma unroll
  for (int nt = 0; nt < 4; ++nt) {
    int d_loc = nt * 16 + (lane & 15);
#pragma unroll
    for (int j = 0; j < 8; ++j) {
      int n_r = wave * 16 + ((lane >> 4) << 3) + j;
      float nv = normS[n_r] + 1e-6f;
      outp[(rowb + n0 + n_r) * 1024 + col + d_loc] =
          (_Float16)(oacc[nt][j] / nv);
    }
  }
}

// ---------------------------------------------------------------------------
// Host-side orchestration
// ---------------------------------------------------------------------------
extern "C" void kernel_launch(void* const* d_in, const int* in_sizes, int n_in,
                              void* d_out, int out_size, void* d_ws,
                              size_t ws_size, hipStream_t stream) {
  const float* x    = (const float*)d_in[0];   // queries [4,4096,1024]
  const float* Wq   = (const float*)d_in[2];
  const float* bq   = (const float*)d_in[3];
  const float* Wk   = (const float*)d_in[4];
  const float* bk   = (const float*)d_in[5];
  const float* Wv   = (const float*)d_in[6];
  const float* bv   = (const float*)d_in[7];
  const float* Wo   = (const float*)d_in[8];
  const float* bo   = (const float*)d_in[9];
  const float* proj = (const float*)d_in[10];  // [64,256]
  float* out = (float*)d_out;

  const size_t MB32 = 33554432ull;  // 16384*1024*2 bytes (f16 activation)
  char* w = (char*)d_ws;
  _Float16* qs   = (_Float16*)(w);                         // scaled q (f16)
  _Float16* ksb  = (_Float16*)(w + MB32);                  // scaled k (f16)
  _Float16* vhb  = (_Float16*)(w + 2 * MB32);              // v (f16); reused as pre-out
  float*    diagq = (float*)(w + 3 * MB32);                // [16384,16]
  float*    diagk = (float*)(w + 3 * MB32 + 1048576);
  float*    kvb   = (float*)(w + 3 * MB32 + 2097152);      // [64,256,64] f32
  float*    ksumb = (float*)(w + 3 * MB32 + 2097152 + 4194304);  // [64,256]

  const int M = 16384, DM = 1024;
  const float r4 = 0.3535533905932738f;  // 1/64^(1/4)
  dim3 blk(256);
  dim3 gG(DM / 128, M / 128);  // (8,128)

  // q/k/v projections (q,k pre-scaled by data_normalizer)
  gemm_kernel<float, _Float16><<<gG, blk, 0, stream>>>(
      x, Wq, bq, qs, M, DM, DM, DM, DM, DM, r4);
  gemm_kernel<float, _Float16><<<gG, blk, 0, stream>>>(
      x, Wk, bk, ksb, M, DM, DM, DM, DM, DM, r4);
  gemm_kernel<float, _Float16><<<gG, blk, 0, stream>>>(
      x, Wv, bv, vhb, M, DM, DM, DM, DM, DM, 1.0f);

  // per-head squared norms of scaled q/k
  diag_kernel<<<1024, blk, 0, stream>>>(qs, diagq);
  diag_kernel<<<1024, blk, 0, stream>>>(ksb, diagk);

  // kv = k'^T v and ksum = sum_n k'  (k' fused/recomputed)
  kv_kernel<<<dim3(2, 64), blk, 0, stream>>>(ksb, vhb, proj, diagk, kvb, ksumb);

  // qkv/norm (q' fused/recomputed); writes pre-output into the v buffer
  qkv_kernel<<<dim3(32, 64), blk, 0, stream>>>(qs, kvb, ksumb, proj, diagq,
                                               vhb);

  // final projection with Wo -> f32 output
  gemm_kernel<_Float16, float><<<gG, blk, 0, stream>>>(
      vhb, Wo, bo, out, M, DM, DM, DM, DM, DM, 1.0f);
}
